// BSpanDecoder_52828097741239
// MI455X (gfx1250) — matrix operand order repaired and checked by reading, not compiled, not run on hardware
//
#include <hip/hip_runtime.h>
#include <hip/hip_bf16.h>

#define T_DIM 256
#define B_DIM 64
#define H_DIM 1024
#define E_DIM 512
#define V_DIM 32000
#define VT    (V_DIM + T_DIM)   /* 32256 */
#define AGG_LD (VT + 1)         /* 32257, +1 = IGN drop bucket */
#define UNK_TOK 2
#define IGN_TOK 0
#define EPS_F 1e-10f

typedef __attribute__((ext_vector_type(16))) __bf16 v16bf;
typedef __attribute__((ext_vector_type(8)))  __bf16 v8bf;
typedef __attribute__((ext_vector_type(4)))  __bf16 v4bf;
typedef __attribute__((ext_vector_type(8)))  float  v8f;

// ---- workspace layout: float region (indices in floats) ----
#define WS_HIDW1   0          // 65536  (B,H)
#define WS_ATT     65536      // 16384  (B,T)
#define WS_ALPHA   81920      // 16384  (B,T)
#define WS_FFIN    98304      // 163840 (B,2560) [ez|ctx|hid]
#define WS_S       262144     // 16384  (B,T)
#define WS_E       278528     // 16384  (B,T)
#define WS_SMAX    294912     // 64
#define WS_TOTAL   294976     // 64
#define WS_MAXAGG  295040     // 64 (uint bits)
#define WS_M       295104     // 64
#define WS_INVZ    295168     // 64
#define WS_CSCALE  295232     // 64
#define WS_BASEC   295296     // 64
#define WS_AGG     295360     // 64*32257 = 2064448
#define WS_FLOATS  2359808
// ---- bf16 region (indices in bf16 elements, base = ws + WS_FLOATS) ----
#define BF_ENC     0L                    // 16384*1024 = 16777216
#define BF_W2      16777216L             // 1024*1024
#define BF_C1      17825792L             // 1024*1024
#define BF_LH      18874368L             // 64*1024
#define BF_ELEMS   18939904L

// A fragment 16x32 (MxK), bf16 source: lane l -> row l&15; k = k0+16h+{0..15}
__device__ __forceinline__ v16bf load_a_frag(const __bf16* __restrict__ base, int ld,
                                             int row0, int k0) {
  int l = threadIdx.x & 31;
  const __bf16* p = base + (size_t)(row0 + (l & 15)) * ld + k0 + 16 * (l >> 4);
  v8bf lo = *(const v8bf*)p;
  v8bf hi = *(const v8bf*)(p + 8);
  return __builtin_shufflevector(lo, hi, 0,1,2,3,4,5,6,7,8,9,10,11,12,13,14,15);
}

// B fragment 32x16 (KxN) from (N,K)-major bf16: lane holds n=l&15, k = k0+16h+{0..15}
__device__ __forceinline__ v16bf load_b_frag(const __bf16* __restrict__ base, int ld,
                                             int n0, int k0) {
  int l = threadIdx.x & 31;
  const __bf16* p = base + (size_t)(n0 + (l & 15)) * ld + k0 + 16 * (l >> 4);
  v8bf lo = *(const v8bf*)p;
  v8bf hi = *(const v8bf*)(p + 8);
  return __builtin_shufflevector(lo, hi, 0,1,2,3,4,5,6,7,8,9,10,11,12,13,14,15);
}

// B fragment with inline f32 -> bf16 (for read-once weights, native cvt)
__device__ __forceinline__ v16bf load_b_frag_f32(const float* __restrict__ base, int ld,
                                                 int n0, int k0) {
  int l = threadIdx.x & 31;
  const float* p = base + (size_t)(n0 + (l & 15)) * ld + k0 + 16 * (l >> 4);
  v16bf f;
#pragma unroll
  for (int i = 0; i < 16; ++i) f[i] = (__bf16)p[i];
  return f;
}

// ---------------------------------------------------------------------------
// f32 -> packed bf16 conversion, fixed 1024 columns, strided source rows
// ---------------------------------------------------------------------------
__global__ void cvt1024_bf16(const float* __restrict__ in, long in_ld, long in_off,
                             __bf16* __restrict__ out, long ngroups) {
  long g = (long)blockIdx.x * blockDim.x + threadIdx.x;
  if (g >= ngroups) return;
  long row = g >> 8;
  int col = (int)(g & 255) * 4;
  const float4 v = *(const float4*)(in + row * in_ld + in_off + col);
  v4bf o;
  o[0] = (__bf16)v.x; o[1] = (__bf16)v.y; o[2] = (__bf16)v.z; o[3] = (__bf16)v.w;
  *(v4bf*)(out + row * 1024 + col) = o;
}

// ---------------------------------------------------------------------------
// Fused: out[b,t] += sum_n post(b,n) * tanh( (enc @ W^T)[r,n] + pre(b,n) )
// ---------------------------------------------------------------------------
__global__ __launch_bounds__(128)
void enc_gemm_tanh_reduce(const __bf16* __restrict__ enc, const __bf16* __restrict__ W,
                          const float* __restrict__ preRow,   // (H)
                          const float* __restrict__ preMat,   // (B,H) or null
                          const float* __restrict__ postRow,  // (H) or null
                          const float* __restrict__ postMat,  // (B,H) or null
                          float* __restrict__ out)            // (B,T) atomic
{
  int wave = threadIdx.x >> 5;
  int lane = threadIdx.x & 31;
  int r0 = blockIdx.x * 64 + wave * 16;   // aligned => all rows share t = blockIdx.x
  int n0 = blockIdx.y * 64;
  v8f acc[4] = {};
#pragma unroll 2
  for (int k0 = 0; k0 < H_DIM; k0 += 32) {
    v16bf a = load_a_frag(enc, H_DIM, r0, k0);
#pragma unroll
    for (int j = 0; j < 4; ++j) {
      v16bf bw = load_b_frag(W, H_DIM, n0 + 16 * j, k0);
      acc[j] = __builtin_amdgcn_wmma_f32_16x16x32_bf16(false, a, false, bw,
                                                       (short)0, acc[j], false, false);
    }
  }
  int h  = lane >> 4;
  int nl = lane & 15;
  int t  = blockIdx.x;
  float partial[8];
#pragma unroll
  for (int v = 0; v < 8; ++v) {
    int b = wave * 16 + v + 8 * h;
    float sum = 0.f;
#pragma unroll
    for (int j = 0; j < 4; ++j) {
      int n = n0 + 16 * j + nl;
      float pre = preRow[n] + (preMat ? preMat[(size_t)b * H_DIM + n] : 0.f);
      float e   = tanhf(acc[j][v] + pre);
      float po  = postRow ? postRow[n] : postMat[(size_t)b * H_DIM + n];
      sum += e * po;
    }
#pragma unroll
    for (int m = 1; m < 16; m <<= 1) sum += __shfl_xor(sum, m, 32);
    partial[v] = sum;
  }
  if (nl == 0) {
#pragma unroll
    for (int v = 0; v < 8; ++v) {
      int b = wave * 16 + v + 8 * h;
      atomicAdd(&out[b * T_DIM + t], partial[v]);
    }
  }
}

// ---------------------------------------------------------------------------
// gen_score = lh @ ffnn_out_w^T + b  :  M=64, N=32000, K=1024 (WMMA)
// ---------------------------------------------------------------------------
__global__ __launch_bounds__(128)
void gen_score_kernel(const __bf16* __restrict__ lh, const float* __restrict__ Wout,
                      const float* __restrict__ bout, float* __restrict__ gen)
{
  int wave = threadIdx.x >> 5;
  int lane = threadIdx.x & 31;
  int n0 = (blockIdx.x * 4 + wave) * 16;
  v8f acc[4] = {};
#pragma unroll 2
  for (int k0 = 0; k0 < H_DIM; k0 += 32) {
    v16bf bw = load_b_frag_f32(Wout, H_DIM, n0, k0);
#pragma unroll
    for (int mt = 0; mt < 4; ++mt) {
      v16bf a = load_a_frag(lh, H_DIM, mt * 16, k0);
      acc[mt] = __builtin_amdgcn_wmma_f32_16x16x32_bf16(false, a, false, bw,
                                                        (short)0, acc[mt], false, false);
    }
  }
  int h = lane >> 4, nl = lane & 15;
  int n = n0 + nl;
  float bb = bout[n];
#pragma unroll
  for (int mt = 0; mt < 4; ++mt)
#pragma unroll
    for (int v = 0; v < 8; ++v) {
      int b = mt * 16 + v + 8 * h;
      gen[(size_t)b * VT + n] = acc[mt][v] + bb;
    }
}

// ---------------------------------------------------------------------------
// small helper kernels
// ---------------------------------------------------------------------------
__global__ void zero_kernel(float* p, long n) {
  for (long i = (long)blockIdx.x * blockDim.x + threadIdx.x; i < n;
       i += (long)gridDim.x * blockDim.x) p[i] = 0.f;
}

__global__ void hidw1_kernel(const float* __restrict__ hid, const float* __restrict__ aw,
                             float* __restrict__ hidW1) {
  int tid = blockIdx.x * blockDim.x + threadIdx.x;     // 65536
  int b = tid >> 10, hh = tid & 1023;
  const float* w = aw + (size_t)hh * 2048;             // W1 = attn_w[:, :H]
  const float* x = hid + (size_t)b * H_DIM;
  float acc = 0.f;
  for (int k = 0; k < H_DIM; ++k) acc += w[k] * x[k];
  hidW1[tid] = acc;
}

__global__ void ez_kernel(const int* __restrict__ z, const float* __restrict__ emb,
                          const float* __restrict__ ecw, const float* __restrict__ ecb,
                          float* __restrict__ ffin) {
  int tid = blockIdx.x * blockDim.x + threadIdx.x;     // 64*512
  int b = tid >> 9, e = tid & 511;
  const float* row = emb + (size_t)z[b] * E_DIM;
  const float* w = ecw + (size_t)e * E_DIM;
  float acc = ecb[e] + row[e];
  for (int k = 0; k < E_DIM; ++k) acc += row[k] * w[k];
  ffin[(size_t)b * 2560 + e] = acc;
}

__global__ __launch_bounds__(256)
void softmax_row_kernel(const float* __restrict__ in, float* __restrict__ out) {
  __shared__ float red[256];
  int b = blockIdx.x, t = threadIdx.x;
  float v = in[b * T_DIM + t];
  red[t] = v; __syncthreads();
  for (int st = 128; st > 0; st >>= 1) { if (t < st) red[t] = fmaxf(red[t], red[t + st]); __syncthreads(); }
  float m = red[0]; __syncthreads();
  float e = expf(v - m);
  red[t] = e; __syncthreads();
  for (int st = 128; st > 0; st >>= 1) { if (t < st) red[t] += red[t + st]; __syncthreads(); }
  out[b * T_DIM + t] = e / red[0];
}

__global__ void ctx_kernel(const float* __restrict__ alpha, const float* __restrict__ enc,
                           const float* __restrict__ hid, float* __restrict__ ffin) {
  int tid = blockIdx.x * blockDim.x + threadIdx.x;     // 65536
  int b = tid >> 10, hh = tid & 1023;
  float acc = 0.f;
  for (int t = 0; t < T_DIM; ++t)
    acc += alpha[b * T_DIM + t] * enc[((size_t)t * B_DIM + b) * H_DIM + hh];
  ffin[(size_t)b * 2560 + 512 + hh]  = acc;
  ffin[(size_t)b * 2560 + 1536 + hh] = hid[(size_t)b * H_DIM + hh];
}

__global__ void lh_kernel(const float* __restrict__ ffin, const float* __restrict__ fhw,
                          const float* __restrict__ fhb, float* __restrict__ lh) {
  int tid = blockIdx.x * blockDim.x + threadIdx.x;     // 65536
  int b = tid >> 10, hh = tid & 1023;
  const float* w = fhw + (size_t)hh * 2560;
  const float* x = ffin + (size_t)b * 2560;
  float acc = fhb[hh];
  for (int k = 0; k < 2560; ++k) acc += w[k] * x[k];
  lh[tid] = acc;
}

__global__ __launch_bounds__(256)
void soft1_kernel(const float* __restrict__ s, float* __restrict__ ebuf,
                  float* __restrict__ smax, float* __restrict__ total) {
  __shared__ float red[256];
  int b = blockIdx.x, t = threadIdx.x;
  float v = s[b * T_DIM + t];
  red[t] = v; __syncthreads();
  for (int st = 128; st > 0; st >>= 1) { if (t < st) red[t] = fmaxf(red[t], red[t + st]); __syncthreads(); }
  float m = red[0]; __syncthreads();
  float e = expf(v - m);
  red[t] = e; __syncthreads();
  for (int st = 128; st > 0; st >>= 1) { if (t < st) red[t] += red[t + st]; __syncthreads(); }
  ebuf[b * T_DIM + t] = e;
  if (t == 0) { smax[b] = m; total[b] = red[0]; }
}

__global__ void scatter_kernel(const int* __restrict__ uin, const float* __restrict__ ebuf,
                               float* __restrict__ agg) {
  int b = blockIdx.x, t = threadIdx.x;
  int w = uin[t * B_DIM + b];
  int col = (w == UNK_TOK) ? (V_DIM + t) : ((w == IGN_TOK) ? VT : w);
  atomicAdd(&agg[(size_t)b * AGG_LD + col], (1.0f - EPS_F) * ebuf[b * T_DIM + t]);
}

__global__ void maxagg_kernel(const int* __restrict__ uin, const float* __restrict__ agg,
                              unsigned* __restrict__ maxagg) {
  int b = blockIdx.x, t = threadIdx.x;
  int w = uin[t * B_DIM + b];
  if (w != IGN_TOK) {
    int col = (w == UNK_TOK) ? (V_DIM + t) : w;
    float a = agg[(size_t)b * AGG_LD + col];
    atomicMax(&maxagg[b], __float_as_uint(a));   // a >= 0 -> uint order == float order
  }
}

__global__ __launch_bounds__(256)
void stats_kernel(const float* __restrict__ gen, const float* __restrict__ smax,
                  const float* __restrict__ total, const unsigned* __restrict__ maxagg,
                  const int* __restrict__ uin, const float* __restrict__ ebuf,
                  float* __restrict__ Mb, float* __restrict__ invZ,
                  float* __restrict__ cScale, float* __restrict__ baseC) {
  __shared__ float red[256];
  int b = blockIdx.x, tid = threadIdx.x;
  const float* g = gen + (size_t)b * VT;
  float m = -3.4e38f;
  for (int j = tid; j < V_DIM; j += 256) m = fmaxf(m, g[j]);
  red[tid] = m; __syncthreads();
  for (int st = 128; st > 0; st >>= 1) { if (tid < st) red[tid] = fmaxf(red[tid], red[tid + st]); __syncthreads(); }
  float genmax = red[0]; __syncthreads();
  float sm = smax[b], tot = total[b];
  float mag = __uint_as_float(maxagg[b]);
  float M = fmaxf(genmax, sm + logf(EPS_F * tot + mag));
  float zg = 0.f;
  for (int j = tid; j < V_DIM; j += 256) zg += expf(g[j] - M);
  red[tid] = zg; __syncthreads();
  for (int st = 128; st > 0; st >>= 1) { if (tid < st) red[tid] += red[tid + st]; __syncthreads(); }
  float Zg = red[0]; __syncthreads();
  int w = uin[tid * B_DIM + b];
  red[tid] = (w != IGN_TOK) ? ebuf[b * T_DIM + tid] : 0.f; __syncthreads();
  for (int st = 128; st > 0; st >>= 1) { if (tid < st) red[tid] += red[tid + st]; __syncthreads(); }
  if (tid == 0) {
    float sumagg = (1.0f - EPS_F) * red[0];
    float es = expf(sm - M);
    float Zc = es * ((float)VT * EPS_F * tot + sumagg);
    float Z = Zg + Zc;
    Mb[b] = M; invZ[b] = 1.f / Z; cScale[b] = es / Z; baseC[b] = EPS_F * tot;
  }
}

__global__ void final_kernel(float* __restrict__ proba, const float* __restrict__ agg,
                             const float* __restrict__ Mb, const float* __restrict__ invZ,
                             const float* __restrict__ cScale, const float* __restrict__ baseC) {
  long tid = (long)blockIdx.x * blockDim.x + threadIdx.x;
  if (tid >= (long)B_DIM * VT) return;
  int b = (int)(tid / VT);
  int j = (int)(tid - (long)b * VT);
  float c = (baseC[b] + agg[(size_t)b * AGG_LD + j]) * cScale[b];
  float o = c;
  if (j < V_DIM) o += expf(proba[tid] - Mb[b]) * invZ[b];   // in-place
  proba[tid] = o;
}

extern "C" void kernel_launch(void* const* d_in, const int* in_sizes, int n_in,
                              void* d_out, int out_size, void* d_ws, size_t ws_size,
                              hipStream_t stream) {
  (void)in_sizes; (void)n_in; (void)out_size; (void)ws_size;
  const float* enc = (const float*)d_in[0];
  const float* hid = (const float*)d_in[1];
  const int*   z   = (const int*)d_in[2];
  const int*   uin = (const int*)d_in[3];
  const float* emb = (const float*)d_in[4];
  const float* ecw = (const float*)d_in[5];
  const float* ecb = (const float*)d_in[6];
  const float* aw  = (const float*)d_in[7];
  const float* ab  = (const float*)d_in[8];
  const float* av  = (const float*)d_in[9];
  const float* fhw = (const float*)d_in[10];
  const float* fhb = (const float*)d_in[11];
  const float* fow = (const float*)d_in[12];
  const float* fob = (const float*)d_in[13];
  const float* c1w = (const float*)d_in[14];
  const float* c1b = (const float*)d_in[15];

  float* outp  = (float*)d_out;
  float* lh    = outp;                       // (1,B,H)
  float* proba = outp + B_DIM * H_DIM;       // (B, V+T); holds gen_score for j<V first

  float* ws     = (float*)d_ws;
  float* hidW1  = ws + WS_HIDW1;
  float* att    = ws + WS_ATT;
  float* alpha  = ws + WS_ALPHA;
  float* ffin   = ws + WS_FFIN;
  float* s      = ws + WS_S;
  float* ebuf   = ws + WS_E;
  float* smaxp  = ws + WS_SMAX;
  float* totalp = ws + WS_TOTAL;
  unsigned* magp = (unsigned*)(ws + WS_MAXAGG);
  float* Mbp    = ws + WS_M;
  float* invZp  = ws + WS_INVZ;
  float* cScp   = ws + WS_CSCALE;
  float* bCp    = ws + WS_BASEC;
  float* agg    = ws + WS_AGG;
  __bf16* bfbase = (__bf16*)(ws + WS_FLOATS);
  __bf16* encbf = bfbase + BF_ENC;
  __bf16* w2bf  = bfbase + BF_W2;
  __bf16* c1bf  = bfbase + BF_C1;
  __bf16* lhbf  = bfbase + BF_LH;

  zero_kernel<<<2048, 256, 0, stream>>>(ws, (long)WS_FLOATS);
  // pre-convert multi-read operands to packed bf16 (cols = 1024)
  cvt1024_bf16<<<16384, 256, 0, stream>>>(enc, 1024, 0, encbf, 4194304L);  // (T*B, H)
  cvt1024_bf16<<<1024, 256, 0, stream>>>(aw, 2048, 1024, w2bf, 262144L);   // W2 = attn_w[:,H:]
  cvt1024_bf16<<<1024, 256, 0, stream>>>(c1w, 1024, 0, c1bf, 262144L);     // copy1_w
  hidw1_kernel<<<256, 256, 0, stream>>>(hid, aw, hidW1);
  // att[b,t] = sum_h attn_v[h] * tanh(enc@W2^T + hidW1 + attn_b)
  enc_gemm_tanh_reduce<<<dim3(256, 16), 128, 0, stream>>>(
      encbf, w2bf, ab, hidW1, av, nullptr, att);
  softmax_row_kernel<<<B_DIM, 256, 0, stream>>>(att, alpha);
  ez_kernel<<<128, 256, 0, stream>>>(z, emb, ecw, ecb, ffin);
  ctx_kernel<<<256, 256, 0, stream>>>(alpha, enc, hid, ffin);
  lh_kernel<<<256, 256, 0, stream>>>(ffin, fhw, fhb, lh);
  cvt1024_bf16<<<64, 256, 0, stream>>>(lh, 1024, 0, lhbf, 16384L);
  gen_score_kernel<<<V_DIM / 64, 128, 0, stream>>>(lhbf, fow, fob, proba);
  // s[b,t] = sum_h lh[b,h] * tanh(enc@copy1_w^T + copy1_b)
  enc_gemm_tanh_reduce<<<dim3(256, 16), 128, 0, stream>>>(
      encbf, c1bf, c1b, nullptr, nullptr, lh, s);
  soft1_kernel<<<B_DIM, 256, 0, stream>>>(s, ebuf, smaxp, totalp);
  scatter_kernel<<<B_DIM, T_DIM, 0, stream>>>(uin, ebuf, agg);
  maxagg_kernel<<<B_DIM, T_DIM, 0, stream>>>(uin, agg, magp);
  stats_kernel<<<B_DIM, 256, 0, stream>>>(proba, smaxp, totalp, magp, uin, ebuf,
                                          Mbp, invZp, cScp, bCp);
  final_kernel<<<(B_DIM * VT + 255) / 256, 256, 0, stream>>>(proba, agg, Mbp, invZp, cScp, bCp);
}
